// OrientedRepPointsHead_5411658793074
// MI455X (gfx1250) — matrix operand order repaired and checked by reading, not compiled, not run on hardware
//
#include <hip/hip_runtime.h>

#define DEVI __device__ __forceinline__

typedef unsigned short bf16s;                                   // bf16 storage
typedef __attribute__((ext_vector_type(16))) __bf16 v16bf;      // WMMA A/B frag
typedef __attribute__((ext_vector_type(8)))  float  v8f;        // WMMA C/D frag

union ABFrag { v16bf v; unsigned int u[8]; };
union CFrag  { v8f  v; float        f[8]; };

#ifndef __has_builtin
#define __has_builtin(x) 0
#endif
#if __has_builtin(__builtin_amdgcn_global_load_async_to_lds_b128) && \
    __has_builtin(__builtin_amdgcn_s_wait_asynccnt)
#define HAVE_ASYNC_LDS 1
#else
#define HAVE_ASYNC_LDS 0
#endif

#if HAVE_ASYNC_LDS
typedef int v4i_raw __attribute__((vector_size(16)));           // GCC-style int4
typedef __attribute__((address_space(1))) v4i_raw* gv4p;        // global AS1
typedef __attribute__((address_space(3))) v4i_raw* lv4p;        // LDS AS3
#define GPTR(p) ((gv4p)(p))
#define LPTR(p) ((lv4p)(p))
#endif

static DEVI float bf2f(bf16s x) {
  unsigned int u = ((unsigned int)x) << 16;
  float f; __builtin_memcpy(&f, &u, 4); return f;
}
static DEVI bf16s f2bf(float f) {
  unsigned int u; __builtin_memcpy(&u, &f, 4);
  u += 0x7FFFu + ((u >> 16) & 1u);               // round-to-nearest-even
  return (bf16s)(u >> 16);
}

// Problem constants
#define BATCH 4
#define CHN   256
#define HH    64
#define WW    64
#define HW    (HH * WW)

// ---------------------------------------------------------------------------
// Layout converters
// ---------------------------------------------------------------------------
__global__ __launch_bounds__(256) void cvt_feat(const float* __restrict__ src,
                                                bf16s* __restrict__ dst) {
  int id = blockIdx.x * 256 + threadIdx.x;       // flat over NHWC (4*64*64*256)
  int c = id & 255;
  int w = (id >> 8) & 63;
  int h = (id >> 14) & 63;
  int b = id >> 20;
  dst[id] = f2bf(src[((size_t)(b * CHN + c) * HH + h) * WW + w]);
}

// OIHW fp32 (256,256,3,3) -> bf16 [tap][co][ci]
__global__ __launch_bounds__(256) void wcvt(const float* __restrict__ src,
                                            bf16s* __restrict__ dst) {
  int id = blockIdx.x * 256 + threadIdx.x;       // 9*256*256
  int ci  = id & 255;
  int co  = (id >> 8) & 255;
  int tap = id >> 16;
  dst[id] = f2bf(src[((size_t)co * 256 + ci) * 9 + tap]);
}

__global__ __launch_bounds__(256) void zero_pad(bf16s* __restrict__ z) {
  z[threadIdx.x] = 0;
}

// ---------------------------------------------------------------------------
// WMMA helpers (ISA 7.12.2 layouts, wave32).  Wave tile: 16 M x 64 N.
// ---------------------------------------------------------------------------
static DEVI void mma_step(const bf16s (*Ac)[32], const bf16s (*Bc)[32],
                          int lane, int wm, int wn, CFrag* acc) {
  const int half8 = (lane >> 4) << 3;            // A: K half offset 0/8
  const int l15   = lane & 15;
  const int mrow  = (wm << 4) + l15;

  ABFrag a;
#pragma unroll
  for (int v = 0; v < 8; ++v) {
    int ka = (v < 4) ? (half8 + (v << 1)) : (16 + half8 + ((v - 4) << 1));
    a.u[v] = *(const unsigned int*)(&Ac[mrow][ka]);
  }
#pragma unroll
  for (int j = 0; j < 4; ++j) {
    ABFrag bf;
    const int ncol = (wn << 6) + (j << 4) + l15;
#pragma unroll
    for (int v = 0; v < 8; ++v)
      bf.u[v] = *(const unsigned int*)(&Bc[ncol][(v < 4) ? (half8 + (v << 1)) : (16 + half8 + ((v - 4) << 1))]);
    acc[j].v = __builtin_amdgcn_wmma_f32_16x16x32_bf16(false, a.v, false, bf.v,
                                                       (short)0, acc[j].v, false, false);
  }
}

static DEVI void store_out4(const CFrag* acc, int lane, int wm, int wn,
                            int row, int nbase, float* outF, bf16s* outB) {
  const int mb  = (wm << 4) + ((lane >= 16) ? 8 : 0);
  const int ncb = nbase + (wn << 6) + (lane & 15);
#pragma unroll
  for (int v = 0; v < 8; ++v) {
    size_t oidx = ((size_t)row * 64 + (mb + v)) * 256;
#pragma unroll
    for (int j = 0; j < 4; ++j) {
      if (outF) outF[oidx + ncb + j * 16] = acc[j].f[v];
      if (outB) outB[oidx + ncb + j * 16] = f2bf(acc[j].f[v]);
    }
  }
}

// ---------------------------------------------------------------------------
// Implicit-GEMM conv3x3 (C->C), bf16 WMMA, fp32 accumulate.
// WG tile: 64 pixels x 128 out-channels; async double-buffered LDS staging.
// grid: (B*H, C/128), block: 256 (8 wave32)
// ---------------------------------------------------------------------------
__global__ __launch_bounds__(256) void conv_wmma(const bf16s* __restrict__ act,
                                                 const bf16s* __restrict__ wt,
                                                 const bf16s* __restrict__ zpad,
                                                 float* __restrict__ outF,
                                                 bf16s* __restrict__ outB) {
  __shared__ bf16s As[2][64][32];
  __shared__ bf16s Bs[2][128][32];
  const int row = blockIdx.x;                    // b*64 + h
  const int b = row >> 6, h = row & 63;
  const int nbase = blockIdx.y << 7;
  const int tid = threadIdx.x, lane = tid & 31, wave = tid >> 5;
  const int wm = wave & 3, wn = wave >> 2;
  CFrag acc[4];
#pragma unroll
  for (int j = 0; j < 4; ++j)
#pragma unroll
    for (int v = 0; v < 8; ++v) acc[j].f[v] = 0.f;

  const int apix = tid >> 2, acg = (tid & 3) << 3;   // A: pixel, 8-ch group
  const int bco  = tid >> 1, bcg = (tid & 1) << 4;   // B: out-ch row, 16-ch group

  auto issue = [&](int s, int buf) {
    const int tap = s >> 3, kc = s & 7;
    const int hs = h + tap / 3 - 1;
    const int wsft = apix + tap % 3 - 1;
    const bool av = (hs >= 0) && (hs < HH) && (wsft >= 0) && (wsft < WW);
    const bf16s* ga = av ? act + ((size_t)((b * HH + hs) * WW + wsft)) * CHN + kc * 32 + acg
                         : zpad;
    const bf16s* gb = wt + ((size_t)(tap * 256 + nbase + bco)) * 256 + kc * 32 + bcg;
#if HAVE_ASYNC_LDS
    __builtin_amdgcn_global_load_async_to_lds_b128(GPTR(ga), LPTR(&As[buf][apix][acg]), 0, 0);
    __builtin_amdgcn_global_load_async_to_lds_b128(GPTR(gb), LPTR(&Bs[buf][bco][bcg]), 0, 0);
    __builtin_amdgcn_global_load_async_to_lds_b128(GPTR(gb + 8), LPTR(&Bs[buf][bco][bcg + 8]), 0, 0);
#else
    uint4 a4 = av ? *(const uint4*)ga : make_uint4(0u, 0u, 0u, 0u);
    *(uint4*)(&As[buf][apix][acg]) = a4;
    *(uint4*)(&Bs[buf][bco][bcg]) = *(const uint4*)gb;
    *(uint4*)(&Bs[buf][bco][bcg + 8]) = *(const uint4*)(gb + 8);
#endif
  };

  issue(0, 0);
  for (int s = 0; s < 72; ++s) {                 // 9 taps * 8 K-chunks of 32
    const int cur = s & 1;
    if (s < 71) issue(s + 1, cur ^ 1);
#if HAVE_ASYNC_LDS
    if (s < 71) __builtin_amdgcn_s_wait_asynccnt(3);  // current buffer complete
    else        __builtin_amdgcn_s_wait_asynccnt(0);
#endif
    __syncthreads();
    mma_step(As[cur], Bs[cur], lane, wm, wn, acc);
    __syncthreads();                             // protect cur before re-fill
  }
  store_out4(acc, lane, wm, wn, row, nbase, outF, outB);
}

// ---------------------------------------------------------------------------
// GroupNorm: reduce (per b,group over 8ch x 4096px) then apply (fp32 -> bf16)
// ---------------------------------------------------------------------------
__global__ __launch_bounds__(256) void gn_reduce(const float* __restrict__ x,
                                                 float* __restrict__ mean,
                                                 float* __restrict__ rstd) {
  const int bg = blockIdx.x;                     // 0..127
  const int b = bg >> 5, g = bg & 31;
  const float* base = x + (size_t)b * HW * CHN + g * 8;
  float s = 0.f, ss = 0.f;
  for (int p = threadIdx.x; p < HW; p += 256) {
    const float* px = base + (size_t)p * CHN;
#pragma unroll
    for (int c = 0; c < 8; ++c) { float v = px[c]; s += v; ss += v * v; }
  }
  __shared__ float rs[256], rq[256];
  rs[threadIdx.x] = s; rq[threadIdx.x] = ss;
  __syncthreads();
  for (int off = 128; off > 0; off >>= 1) {
    if (threadIdx.x < off) { rs[threadIdx.x] += rs[threadIdx.x + off];
                             rq[threadIdx.x] += rq[threadIdx.x + off]; }
    __syncthreads();
  }
  if (threadIdx.x == 0) {
    float m = rs[0] * (1.0f / 32768.0f);
    float v = rq[0] * (1.0f / 32768.0f) - m * m;
    mean[bg] = m;
    rstd[bg] = rsqrtf(v + 1e-5f);
  }
}

__global__ __launch_bounds__(256) void gn_apply(const float* __restrict__ x,
                                                const float* __restrict__ mean,
                                                const float* __restrict__ rstd,
                                                const float* __restrict__ gamma,
                                                const float* __restrict__ beta,
                                                bf16s* __restrict__ out) {
  int id = blockIdx.x * 256 + threadIdx.x;       // NHWC flat
  int c = id & 255;
  int b = id >> 20;
  int bg = b * 32 + (c >> 3);
  float y = (x[id] - mean[bg]) * rstd[bg] * gamma[c] + beta[c];
  out[id] = f2bf(y);
}

// ---------------------------------------------------------------------------
// Small-output conv3x3 (Co = 18 or 15), bf16 NHWC in, fp32 NCHW out
// ---------------------------------------------------------------------------
__global__ __launch_bounds__(256) void small_conv(const bf16s* __restrict__ act,
                                                  const float* __restrict__ wgt,
                                                  float* __restrict__ out, int Co) {
  int id = blockIdx.x * 256 + threadIdx.x;       // B*Co*HW
  int w = id & 63;
  int h = (id >> 6) & 63;
  int co = (id >> 12) % Co;
  int b = id / (Co * HW);
  float acc = 0.f;
  for (int kh = 0; kh < 3; ++kh) {
    int hs = h + kh - 1;
    if (hs < 0 || hs >= HH) continue;
    for (int kw = 0; kw < 3; ++kw) {
      int wsft = w + kw - 1;
      if (wsft < 0 || wsft >= WW) continue;
      const bf16s* ap = act + ((size_t)((b * HH + hs) * WW + wsft)) * CHN;
      const float* wp = wgt + (size_t)co * 2304 + (kh * 3 + kw);
      for (int c = 0; c < CHN; ++c) acc += bf2f(ap[c]) * wp[c * 9];
    }
  }
  out[id] = acc;
}

// ---------------------------------------------------------------------------
// Deformable conv: bilinear coefficients (shared across channels), fp32
// ---------------------------------------------------------------------------
__global__ __launch_bounds__(256) void deform_coeff(const float* __restrict__ off1,
                                                    float4* __restrict__ cw,
                                                    ushort4* __restrict__ cix) {
  int id = blockIdx.x * 256 + threadIdx.x;       // B*9*HW
  int w = id & 63;
  int h = (id >> 6) & 63;
  int k = (id >> 12) % 9;
  int b = id / (9 * HW);
  float dy = off1[(((size_t)(b * 18 + 2 * k) * HH + h) << 6) + w];
  float dx = off1[(((size_t)(b * 18 + 2 * k + 1) * HH + h) << 6) + w];
  float y = dy + (float)(h - 1 + k / 3);
  float x = dx + (float)(w - 1 + k % 3);
  float y0f = floorf(y), x0f = floorf(x);
  float wy = y - y0f, wx = x - x0f;
  int y0 = (int)y0f, x0 = (int)x0f;
  bool vy0 = (y0 >= 0) && (y0 < HH), vy1 = (y0 + 1 >= 0) && (y0 + 1 < HH);
  bool vx0 = (x0 >= 0) && (x0 < WW), vx1 = (x0 + 1 >= 0) && (x0 + 1 < WW);
  float4 c;
  c.x = (vy0 && vx0) ? (1.f - wy) * (1.f - wx) : 0.f;
  c.y = (vy0 && vx1) ? (1.f - wy) * wx : 0.f;
  c.z = (vy1 && vx0) ? wy * (1.f - wx) : 0.f;
  c.w = (vy1 && vx1) ? wy * wx : 0.f;
  cw[id] = c;
  int cy0 = min(max(y0, 0), HH - 1), cy1 = min(max(y0 + 1, 0), HH - 1);
  int cx0 = min(max(x0, 0), WW - 1), cx1 = min(max(x0 + 1, 0), WW - 1);
  cix[id] = make_ushort4((unsigned short)cy0, (unsigned short)cy1,
                         (unsigned short)cx0, (unsigned short)cx1);
}

static DEVI unsigned int bilerp2(unsigned int q00, unsigned int q01,
                                 unsigned int q10, unsigned int q11, float4 cv) {
  float lo = cv.x * bf2f((bf16s)(q00 & 0xffffu)) + cv.y * bf2f((bf16s)(q01 & 0xffffu))
           + cv.z * bf2f((bf16s)(q10 & 0xffffu)) + cv.w * bf2f((bf16s)(q11 & 0xffffu));
  float hi = cv.x * bf2f((bf16s)(q00 >> 16)) + cv.y * bf2f((bf16s)(q01 >> 16))
           + cv.z * bf2f((bf16s)(q10 >> 16)) + cv.w * bf2f((bf16s)(q11 >> 16));
  return (unsigned int)f2bf(lo) | ((unsigned int)f2bf(hi) << 16);
}

// Deformable conv as WMMA implicit GEMM: A-tile = bilinear-gathered bf16.
// WG tile: 64 pixels x 128 out-channels; grid: (B*H, C/128)
__global__ __launch_bounds__(256) void deform_conv_wmma(const bf16s* __restrict__ act,
                                                        const bf16s* __restrict__ wt,
                                                        const float4* __restrict__ cw,
                                                        const ushort4* __restrict__ cix,
                                                        bf16s* __restrict__ outB) {
  __shared__ bf16s As[64][32];
  __shared__ bf16s Bs[128][32];
  const int row = blockIdx.x;                    // b*64 + h
  const int b = row >> 6, h = row & 63;
  const int nbase = blockIdx.y << 7;
  const int tid = threadIdx.x, lane = tid & 31, wave = tid >> 5;
  const int wm = wave & 3, wn = wave >> 2;
  CFrag acc[4];
#pragma unroll
  for (int j = 0; j < 4; ++j)
#pragma unroll
    for (int v = 0; v < 8; ++v) acc[j].f[v] = 0.f;

  const int apix = tid >> 2, acg = (tid & 3) << 3;
  const int bco  = tid >> 1, bcg = (tid & 1) << 4;

  for (int tap = 0; tap < 9; ++tap) {
    const int cid = (b * 9 + tap) * HW + (h << 6) + apix;
    const float4  cv = cw[cid];
    const ushort4 ix = cix[cid];
    const bf16s* p00 = act + ((size_t)((b * HH + ix.x) * WW + ix.z)) * CHN;
    const bf16s* p01 = act + ((size_t)((b * HH + ix.x) * WW + ix.w)) * CHN;
    const bf16s* p10 = act + ((size_t)((b * HH + ix.y) * WW + ix.z)) * CHN;
    const bf16s* p11 = act + ((size_t)((b * HH + ix.y) * WW + ix.w)) * CHN;
    const bf16s* brow = wt + ((size_t)(tap * 256 + nbase + bco)) * 256;

    for (int kc = 0; kc < 8; ++kc) {
      const int cofs = kc * 32 + acg;
      uint4 q00 = *(const uint4*)(p00 + cofs);
      uint4 q01 = *(const uint4*)(p01 + cofs);
      uint4 q10 = *(const uint4*)(p10 + cofs);
      uint4 q11 = *(const uint4*)(p11 + cofs);
      uint4 av;
      av.x = bilerp2(q00.x, q01.x, q10.x, q11.x, cv);
      av.y = bilerp2(q00.y, q01.y, q10.y, q11.y, cv);
      av.z = bilerp2(q00.z, q01.z, q10.z, q11.z, cv);
      av.w = bilerp2(q00.w, q01.w, q10.w, q11.w, cv);
      *(uint4*)(&As[apix][acg]) = av;
      const int bofs = kc * 32 + bcg;
      *(uint4*)(&Bs[bco][bcg]) = *(const uint4*)(brow + bofs);
      *(uint4*)(&Bs[bco][bcg + 8]) = *(const uint4*)(brow + bofs + 8);
      __syncthreads();
      mma_step(As, Bs, lane, wm, wn, acc);
      __syncthreads();
    }
  }
  store_out4(acc, lane, wm, wn, row, nbase, (float*)nullptr, outB);
}

// ---------------------------------------------------------------------------
// rp1 = rp_init + offset1 ; rp2 = rp1 + offset2   (fp32, NCHW [B,18,H,W])
// ---------------------------------------------------------------------------
__global__ __launch_bounds__(256) void add_off(const float* __restrict__ off1,
                                               const float* __restrict__ off2,
                                               float* __restrict__ out) {
  int id = blockIdx.x * 256 + threadIdx.x;       // B*18*HW
  int w = id & 63;
  int h = (id >> 6) & 63;
  int ch = (id >> 12) % 18;
  int k = ch >> 1;
  float init = (ch & 1) ? (float)(w + k % 3 - 1) : (float)(h + k / 3 - 1);
  float r1 = init + off1[id];
  float r2 = r1 + off2[id];
  out[id] = r1;
  out[294912 + id] = r2;
}

// ---------------------------------------------------------------------------
// Host orchestration
// ---------------------------------------------------------------------------
extern "C" void kernel_launch(void* const* d_in, const int* in_sizes, int n_in,
                              void* d_out, int out_size, void* d_ws, size_t ws_size,
                              hipStream_t stream) {
  (void)in_sizes; (void)n_in; (void)out_size; (void)ws_size;
  const float* feature = (const float*)d_in[0];
  const float* loc_ws  = (const float*)d_in[1];
  const float* loc_g   = (const float*)d_in[2];
  const float* loc_b   = (const float*)d_in[3];
  const float* cls_ws  = (const float*)d_in[4];
  const float* cls_g   = (const float*)d_in[5];
  const float* cls_b   = (const float*)d_in[6];
  const float* pi_w    = (const float*)d_in[7];
  const float* pio_w   = (const float*)d_in[8];
  const float* prd_w   = (const float*)d_in[9];
  const float* pro_w   = (const float*)d_in[10];
  const float* cd_w    = (const float*)d_in[11];
  const float* co_w    = (const float*)d_in[12];
  float* out = (float*)d_out;

  char* ws = (char*)d_ws;
  const size_t ACT_B = (size_t)BATCH * HW * CHN * 2;    // 8,388,608
  bf16s* featA = (bf16s*)(ws);                          // bf16 NHWC feature
  bf16s* B1    = (bf16s*)(ws + ACT_B);
  bf16s* B2    = (bf16s*)(ws + 2 * ACT_B);
  bf16s* B3    = (bf16s*)(ws + 3 * ACT_B);
  float* convF = (float*)(ws + 4 * ACT_B);              // 16,777,216
  bf16s* wbf   = (bf16s*)(ws + 4 * ACT_B + 16777216);   // 9 slots * 589824 bf16
  const size_t WSLOT = 9 * 256 * 256;                   // elems per slot
  char*  p = ws + 4 * ACT_B + 16777216 + 9 * WSLOT * 2; // 60,948,480
  float*   off1 = (float*)p;            p += 1179648;
  float*   off2 = (float*)p;            p += 1179648;
  float4*  cwv  = (float4*)p;           p += 2359296;
  ushort4* cixv = (ushort4*)p;          p += 1179648;
  float*   gmean = (float*)p;           p += 512;
  float*   grstd = (float*)p;           p += 512;
  bf16s*   zpad  = (bf16s*)p;                           // 512B zero pad

  // 0. zero pad block for async zero-fill redirects
  zero_pad<<<1, 256, 0, stream>>>(zpad);

  // 1. Convert feature + all 256->256 weights to bf16 WMMA layouts
  cvt_feat<<<16384, 256, 0, stream>>>(feature, featA);
  const size_t WNUM = 256 * 256 * 9;                    // fp32 elems per conv
  for (int i = 0; i < 3; ++i)
    wcvt<<<2304, 256, 0, stream>>>(loc_ws + (size_t)i * WNUM, wbf + (size_t)i * WSLOT);
  for (int i = 0; i < 3; ++i)
    wcvt<<<2304, 256, 0, stream>>>(cls_ws + (size_t)i * WNUM, wbf + (size_t)(3 + i) * WSLOT);
  wcvt<<<2304, 256, 0, stream>>>(pi_w,  wbf + 6 * WSLOT);
  wcvt<<<2304, 256, 0, stream>>>(prd_w, wbf + 7 * WSLOT);
  wcvt<<<2304, 256, 0, stream>>>(cd_w,  wbf + 8 * WSLOT);

  dim3 cgrid(BATCH * HH, CHN / 128);
  auto convgn = [&](const bf16s* in, int wslot, const float* g, const float* be, bf16s* ob) {
    conv_wmma<<<cgrid, 256, 0, stream>>>(in, wbf + (size_t)wslot * WSLOT, zpad,
                                         convF, (bf16s*)nullptr);
    gn_reduce<<<128, 256, 0, stream>>>(convF, gmean, grstd);
    gn_apply<<<16384, 256, 0, stream>>>(convF, gmean, grstd, g, be, ob);
  };

  // 2. loc subnet: featA -> B1 -> B2 -> B1
  convgn(featA, 0, loc_g,       loc_b,       B1);
  convgn(B1,    1, loc_g + 256, loc_b + 256, B2);
  convgn(B2,    2, loc_g + 512, loc_b + 512, B1);        // loc features in B1

  // 3. offset1 = pio(pi(loc))
  conv_wmma<<<cgrid, 256, 0, stream>>>(B1, wbf + 6 * WSLOT, zpad, (float*)nullptr, B2);
  small_conv<<<1152, 256, 0, stream>>>(B2, pio_w, off1, 18);

  // 4. deform sampling coefficients (shared by both deform convs)
  deform_coeff<<<576, 256, 0, stream>>>(off1, cwv, cixv);

  // 5. offset2 = pro(deform(loc, offset1, prd_w))
  deform_conv_wmma<<<cgrid, 256, 0, stream>>>(B1, wbf + 7 * WSLOT, cwv, cixv, B2);
  small_conv<<<1152, 256, 0, stream>>>(B2, pro_w, off2, 18);

  // 6. cls subnet: featA -> B3 -> B2 -> B3
  convgn(featA, 3, cls_g,       cls_b,       B3);
  convgn(B3,    4, cls_g + 256, cls_b + 256, B2);
  convgn(B2,    5, cls_g + 512, cls_b + 512, B3);        // cls features in B3

  // 7. classification = co(deform(cls, offset1, cd_w)) -> d_out[2*294912..]
  deform_conv_wmma<<<cgrid, 256, 0, stream>>>(B3, wbf + 8 * WSLOT, cwv, cixv, B2);
  small_conv<<<960, 256, 0, stream>>>(B2, co_w, out + 2 * 294912, 15);

  // 8. rp1 / rp2 -> d_out[0..], d_out[294912..]
  add_off<<<1152, 256, 0, stream>>>(off1, off2, out);
}
